// S_GNN_38732015075466
// MI455X (gfx1250) — matrix operand order repaired and checked by reading, not compiled, not run on hardware
//
#include <hip/hip_runtime.h>

// ---------------- types for WMMA fragments ----------------
typedef __bf16 bf16_t;
typedef bf16_t v16bf __attribute__((ext_vector_type(16)));
typedef float  v8f   __attribute__((ext_vector_type(8)));
typedef unsigned int v4u __attribute__((ext_vector_type(4)));

union Frag {
    v16bf v;
    v4u   u[2];
    unsigned short h[16];
};

#define DEV __device__ __forceinline__

DEV unsigned short f2bf(float f) {
    unsigned int u = __float_as_uint(f);
    u += 0x7fffu + ((u >> 16) & 1u);   // round-to-nearest-even
    return (unsigned short)(u >> 16);
}
DEV float bf2f(unsigned short h) {
    return __uint_as_float((unsigned int)h << 16);
}
DEV int imin(int a, int b) { return a < b ? a : b; }

// ---------------- constants ----------------
#define BATCH 512
#define NN    50
#define FEAT  300
#define LD    2048
#define D1    600
#define D1P   640            // K-pad, multiple of 64
#define D2    1200
#define D2P   1216
#define ROWS  25600          // BATCH*NN
#define KC    64             // k-chunk staged to LDS
#define BROW  72             // padded LDS row stride (elements) -> bank-friendly

// =====================================================================
// f32 -> bf16 conversion with K padding (weights + l)
// =====================================================================
__global__ __launch_bounds__(256)
void cvt_kernel(const float* __restrict__ src, unsigned short* __restrict__ dst,
                int rows, int K, int Kpad) {
    int i = blockIdx.x * 256 + threadIdx.x;
    int total = rows * Kpad;
    if (i >= total) return;
    int r = i / Kpad, k = i - r * Kpad;
    dst[i] = (k < K) ? f2bf(src[(size_t)r * K + k]) : (unsigned short)0;
}

// =====================================================================
// bb features + bb FC (K=10, scalar) -> s_bb(bf16, padded 640) and
// left half of combined (cols 0..599), plus combined pad cols 1200..1215
// =====================================================================
__global__ __launch_bounds__(256)
void prep_kernel(const float* __restrict__ s, const float* __restrict__ ps,
                 const float* __restrict__ bb_w, const float* __restrict__ bb_b,
                 unsigned short* __restrict__ s_bb, unsigned short* __restrict__ combined) {
    int row = blockIdx.x;                       // 0..25599
    const float* p = ps + (size_t)row * 4;
    float p0 = p[0], p1 = p[1], p2 = p[2], p3 = p[3];
    float w = p2 - p0, h = p3 - p1;
    float cx = p0 + 0.5f * w, cy = p1 + 0.5f * h;
    float area = w * h, shape = w / (h + 1e-14f);
    float f[10] = {p0, p1, p2, p3, w, h, cx, cy, area, shape};

    for (int j = threadIdx.x; j < D1P; j += 256) {
        float v;
        if (j < FEAT) {
            v = s[(size_t)row * FEAT + j];
        } else if (j < D1) {
            int o = j - FEAT;
            const float* wr = bb_w + o * 10;
            float acc = bb_b[o];
            #pragma unroll
            for (int k = 0; k < 10; k++) acc += f[k] * wr[k];
            v = acc > 0.f ? acc : 0.f;
        } else {
            v = 0.f;
        }
        unsigned short bv = f2bf(v);
        s_bb[(size_t)row * D1P + j] = bv;
        if (j < D1) combined[(size_t)row * D2P + j] = bv;
    }
    for (int j = D2 + threadIdx.x; j < D2P; j += 256)
        combined[(size_t)row * D2P + j] = 0;   // zero pad cols 1200..1215
}

// =====================================================================
// Block-cooperative WMMA GEMM: Y = epilogue(relu(A[M,K] @ W[N,K]^T + b))
// Block (8 waves) computes 128(M) x 64(N). B tile staged to LDS in
// 64x64 k-chunks, double-buffered via GLOBAL_LOAD_ASYNC_TO_LDS_B128
// (ASYNCcnt). A fragments register double-buffered across chunks +
// prefetched. K must be a multiple of 64, M a multiple of 128.
// EPI: 0=bf16 store(relu)  1=bf16 store(relu * lp[b,col])
//      2=f32 store(relu)   3=f32 masked store w/ colOff (out FC)
// =====================================================================
template <int EPI>
__global__ __launch_bounds__(256)
void gemm_kernel(const unsigned short* __restrict__ A, int lda,
                 const unsigned short* __restrict__ W, int ldw,
                 const float* __restrict__ bias,
                 int M, int Nreal, int K,
                 float* __restrict__ outF, unsigned short* __restrict__ outB, int ldo,
                 int colOff, const float* __restrict__ lp, const int* __restrict__ msk) {
    __shared__ unsigned short Bs[2][64 * BROW];

    const int tilesN = (Nreal + 63) >> 6;
    const int bm = blockIdx.x / tilesN;
    const int bn = blockIdx.x - bm * tilesN;
    const int tid = threadIdx.x;
    const int lane = tid & 31;
    const int wv = tid >> 5;
    const int n0 = bn * 64;

    // ---- async stage of one 64(N) x 64(K) chunk of W into LDS buffer ----
    auto stage = [&](int buf, int kc) {
        #pragma unroll
        for (int half = 0; half < 2; half++) {
            int piece = tid + half * 256;            // 512 pieces of 8 elements
            int row = piece >> 3, seg = piece & 7;
            int n = imin(n0 + row, Nreal - 1);
            const unsigned short* g = W + (size_t)n * ldw + kc + seg * 8;
            unsigned int ldsOff = (unsigned int)(size_t)(&Bs[buf][row * BROW + seg * 8]);
            unsigned long long ga = (unsigned long long)(size_t)g;
            asm volatile("global_load_async_to_lds_b128 %0, %1, off"
                         :: "v"(ldsOff), "v"(ga) : "memory");
        }
    };

    // ---- per-wave compute setup: 16 rows x 64 cols ----
    const int aK = (lane >> 4) * 8;
    const int kb = (lane >> 4) * 8;
    const unsigned short* aPtr =
        A + (size_t)(bm * 128 + wv * 16 + (lane & 15)) * lda;
    v8f acc[4] = {};
    const int nChunks = K / KC;

    auto loadApair = [&](Frag& f0, Frag& f1, int kc) {
        f0.u[0] = *(const v4u*)(aPtr + kc + aK);
        f0.u[1] = *(const v4u*)(aPtr + kc + aK + 16);
        f1.u[0] = *(const v4u*)(aPtr + kc + 32 + aK);
        f1.u[1] = *(const v4u*)(aPtr + kc + 32 + aK + 16);
    };
    auto mmaChunk = [&](Frag& f0, Frag& f1, const unsigned short* bufp) {
        #pragma unroll
        for (int kkl = 0; kkl < 2; kkl++) {
            #pragma unroll
            for (int s2 = 0; s2 < 4; s2++) {
                const unsigned short* bp =
                    bufp + (s2 * 16 + (lane & 15)) * BROW + kkl * 32 + kb;
                Frag b;
                b.u[0] = *(const v4u*)bp;               // ds_load_b128
                b.u[1] = *(const v4u*)(bp + 16);
                acc[s2] = __builtin_amdgcn_wmma_f32_16x16x32_bf16(
                    false, (kkl ? f1.v : f0.v), false, b.v, (short)0,
                    acc[s2], false, false);
            }
        }
    };
    auto syncStage = [&](int c) {                       // after computing chunk c
        __syncthreads();                                // everyone done with buf[c&1]
        if (c + 2 < nChunks) {
            stage(c & 1, (c + 2) * KC);
            asm volatile("s_wait_asynccnt 0x2" ::: "memory");  // buf[(c+1)&1] ready
        } else {
            asm volatile("s_wait_asynccnt 0x0" ::: "memory");
        }
        __syncthreads();
    };

    stage(0, 0);
    stage(1, KC);
    Frag aE0, aE1, aO0, aO1;
    loadApair(aE0, aE1, 0);
    asm volatile("s_wait_asynccnt 0x2" ::: "memory");   // buffer 0 landed
    __syncthreads();

    for (int c = 0; c < nChunks; c += 2) {
        if (c + 1 < nChunks) loadApair(aO0, aO1, (c + 1) * KC);
        __builtin_prefetch(aPtr + (c + 2) * KC, 0, 1);  // global_prefetch_b8
        mmaChunk(aE0, aE1, &Bs[0][0]);
        syncStage(c);
        if (c + 1 >= nChunks) break;
        if (c + 2 < nChunks) loadApair(aE0, aE1, (c + 2) * KC);
        mmaChunk(aO0, aO1, &Bs[1][0]);
        syncStage(c + 1);
    }

    // ---- epilogue ----
    const int rbase = (lane >> 4) * 8;
    #pragma unroll
    for (int s2 = 0; s2 < 4; s2++) {
        int col = n0 + s2 * 16 + (lane & 15);
        float bv = (col < Nreal) ? bias[col] : 0.f;
        #pragma unroll
        for (int r = 0; r < 8; r++) {
            int row = bm * 128 + wv * 16 + rbase + r;
            float v = acc[s2][r] + bv;
            v = v > 0.f ? v : 0.f;
            if (EPI == 1) {
                int bidx = row / 50;
                v *= lp[(size_t)bidx * D2 + (col < D2 ? col : 0)];
            }
            if (EPI == 0 || EPI == 1) {
                outB[(size_t)row * ldo + col] = (col < Nreal) ? f2bf(v) : (unsigned short)0;
            } else if (EPI == 2) {
                if (col < Nreal) outF[(size_t)row * ldo + col] = v;
            } else if (EPI == 3) {
                if (col < Nreal) {
                    int bidx = row / 50, n = row - bidx * 50;
                    float mv = (n < msk[bidx]) ? v : 0.f;
                    outF[(size_t)row * ldo + colOff + col] = mv;
                }
            }
        }
    }
}

// =====================================================================
// combined[:,600+j] = t1[:,j] * t2[:,j]   (t1/t2 already relu'd bf16)
// =====================================================================
__global__ __launch_bounds__(256)
void mul_kernel(const unsigned short* __restrict__ t1,
                const unsigned short* __restrict__ t2,
                unsigned short* __restrict__ combined) {
    int i = blockIdx.x * 256 + threadIdx.x;
    if (i >= ROWS * D1) return;
    int row = i / D1, j = i - row * D1;
    float v = bf2f(t1[(size_t)row * D1P + j]) * bf2f(t2[(size_t)row * D1P + j]);
    combined[(size_t)row * D2P + D1 + j] = f2bf(v);
}

// =====================================================================
// adj_raw[b,n,m] = fa4r[b,n,:] . lms[b,m,:]    (per-batch, K=1216)
// =====================================================================
__global__ __launch_bounds__(256)
void gemm_adj_kernel(const unsigned short* __restrict__ fa4r,
                     const unsigned short* __restrict__ lms,
                     float* __restrict__ adj_raw) {
    const int lane = threadIdx.x & 31;
    const int wave = blockIdx.x * 8 + (threadIdx.x >> 5);
    if (wave >= BATCH * 4) return;
    const int b = wave >> 2, tm = wave & 3;

    int aRowG = imin(b * NN + tm * 16 + (lane & 15), ROWS - 1);
    const int aK = (lane >> 4) * 8;
    const int kb = (lane >> 4) * 8;
    const unsigned short* aPtr = fa4r + (size_t)aRowG * D2P;

    const unsigned short* bPtr[4];
    #pragma unroll
    for (int s2 = 0; s2 < 4; s2++) {
        int m = imin(s2 * 16 + (lane & 15), NN - 1);
        bPtr[s2] = lms + (size_t)(b * NN + m) * D2P + kb;
    }

    v8f acc[4] = {};
    auto loadA = [&](Frag& fa, int kk) {
        fa.u[0] = *(const v4u*)(aPtr + kk + aK);
        fa.u[1] = *(const v4u*)(aPtr + kk + aK + 16);
    };
    auto loadB = [&](Frag (&fb)[4], int kk) {
        #pragma unroll
        for (int s2 = 0; s2 < 4; s2++) {
            fb[s2].u[0] = *(const v4u*)(bPtr[s2] + kk);
            fb[s2].u[1] = *(const v4u*)(bPtr[s2] + kk + 16);
        }
    };
    auto mma = [&](Frag& fa, Frag (&fb)[4]) {
        #pragma unroll
        for (int s2 = 0; s2 < 4; s2++)
            acc[s2] = __builtin_amdgcn_wmma_f32_16x16x32_bf16(
                false, fa.v, false, fb[s2].v, (short)0, acc[s2], false, false);
    };

    Frag aE, aO, bE[4], bO[4];
    loadA(aE, 0); loadB(bE, 0);
    int kk = 0;
    for (; kk + 64 <= D2P; kk += 64) {
        loadA(aO, kk + 32); loadB(bO, kk + 32);
        mma(aE, bE);
        if (kk + 64 < D2P) { loadA(aE, kk + 64); loadB(bE, kk + 64); }
        mma(aO, bO);
    }
    if (kk < D2P) mma(aE, bE);

    const int rbase = (lane >> 4) * 8;
    #pragma unroll
    for (int s2 = 0; s2 < 4; s2++) {
        int m = s2 * 16 + (lane & 15);
        #pragma unroll
        for (int r = 0; r < 8; r++) {
            int n = tm * 16 + rbase + r;
            if (n < NN && m < NN)
                adj_raw[((size_t)b * NN + n) * NN + m] = acc[s2][r];
        }
    }
}

// =====================================================================
// masked softmax; writes fp32 adj to d_out and zero-padded bf16 [64x64]
// =====================================================================
__global__ __launch_bounds__(64)
void softmax_kernel(const float* __restrict__ adj_raw, const int* __restrict__ mask_s,
                    float* __restrict__ adj_out, unsigned short* __restrict__ adjb) {
    const int b = blockIdx.x >> 6;
    const int n = blockIdx.x & 63;
    const int m = threadIdx.x;
    if (n >= NN) { adjb[(size_t)b * 4096 + n * 64 + m] = 0; return; }

    const int msk = mask_s[b];
    const float NEG = -__builtin_inff();
    float x = NEG;
    if (m < NN) {
        int mx = n > m ? n : m;
        float bias = ((mx < msk) || (n >= msk)) ? 0.f : NEG;
        if (m == n) bias = NEG;
        if (n == 0 && m == 0 && msk == 1) bias = 0.f;
        x = adj_raw[((size_t)b * NN + n) * NN + m] + bias;
    }

    __shared__ float sm[64];
    sm[m] = x;
    __syncthreads();
    for (int off = 32; off > 0; off >>= 1) {
        if (m < off) sm[m] = fmaxf(sm[m], sm[m + off]);
        __syncthreads();
    }
    float rowmax = sm[0];
    __syncthreads();
    float e = expf(x - rowmax);           // exp(-inf - max) = 0
    sm[m] = e;
    __syncthreads();
    for (int off = 32; off > 0; off >>= 1) {
        if (m < off) sm[m] += sm[m + off];
        __syncthreads();
    }
    float p = e / sm[0];

    if (m < NN) adj_out[((size_t)b * NN + n) * NN + m] = p;
    adjb[(size_t)b * 4096 + n * 64 + m] = (m < NN) ? f2bf(p) : (unsigned short)0;
}

// =====================================================================
// mp[b,n,:] = adj[b,n,:] @ prepared[b,:,:]   (per-batch, K=64 padded)
// =====================================================================
__global__ __launch_bounds__(256)
void gemm_msg_kernel(const unsigned short* __restrict__ adjb,
                     const unsigned short* __restrict__ prepared,
                     unsigned short* __restrict__ mp) {
    const int lane = threadIdx.x & 31;
    const int wave = blockIdx.x * 8 + (threadIdx.x >> 5);
    if (wave >= BATCH * 4 * 19) return;
    const int b = wave / 76;
    const int rem = wave - b * 76;
    const int tm = rem / 19, tn = rem - tm * 19;

    const int nLocal = tm * 16 + (lane & 15);
    const unsigned short* aPtr = adjb + (size_t)b * 4096 + nLocal * 64;
    const int aK = (lane >> 4) * 8;
    const int kb = (lane >> 4) * 16;

    int dIdx[4];
    #pragma unroll
    for (int s2 = 0; s2 < 4; s2++) dIdx[s2] = tn * 64 + s2 * 16 + (lane & 15);

    v8f acc[4] = {};
    #pragma unroll
    for (int kk = 0; kk < 64; kk += 32) {
        Frag a;
        a.u[0] = *(const v4u*)(aPtr + kk + aK);
        a.u[1] = *(const v4u*)(aPtr + kk + aK + 16);
        #pragma unroll
        for (int s2 = 0; s2 < 4; s2++) {
            Frag c;
            #pragma unroll
            for (int j = 0; j < 16; j++) {
                int mrow = imin(kk + aK + (j & 7) + (j >> 3) * 16, NN - 1);
                c.h[j] = prepared[(size_t)(b * NN + mrow) * D2P + dIdx[s2]];
            }
            acc[s2] = __builtin_amdgcn_wmma_f32_16x16x32_bf16(
                false, a.v, false, c.v, (short)0, acc[s2], false, false);
        }
    }

    const int rbase = (lane >> 4) * 8;
    #pragma unroll
    for (int s2 = 0; s2 < 4; s2++) {
        #pragma unroll
        for (int r = 0; r < 8; r++) {
            int n = tm * 16 + rbase + r;
            if (n < NN)
                mp[(size_t)(b * NN + n) * D2P + dIdx[s2]] = f2bf(acc[s2][r]);
        }
    }
}

// =====================================================================
// s_out[:, :, 0:300] = s * mask
// =====================================================================
__global__ __launch_bounds__(256)
void write_s_kernel(const float* __restrict__ s, const int* __restrict__ mask_s,
                    float* __restrict__ out) {
    int i = blockIdx.x * 256 + threadIdx.x;
    if (i >= ROWS * FEAT) return;
    int row = i / FEAT, j = i - row * FEAT;
    int b = row / NN, n = row - b * NN;
    out[(size_t)row * (2 * FEAT) + j] = (n < mask_s[b]) ? s[i] : 0.f;
}

// =====================================================================
// host launcher
// =====================================================================
extern "C" void kernel_launch(void* const* d_in, const int* in_sizes, int n_in,
                              void* d_out, int out_size, void* d_ws, size_t ws_size,
                              hipStream_t stream) {
    const float* l      = (const float*)d_in[0];
    const float* s      = (const float*)d_in[1];
    const float* ps     = (const float*)d_in[2];
    const int*   mask_s = (const int*)  d_in[3];
    const float* bb_w   = (const float*)d_in[4];
    const float* bb_b   = (const float*)d_in[5];
    const float* fa1_w  = (const float*)d_in[6];
    const float* fa1_b  = (const float*)d_in[7];
    const float* fa2_w  = (const float*)d_in[8];
    const float* fa2_b  = (const float*)d_in[9];
    const float* fa3_w  = (const float*)d_in[10];
    const float* fa3_b  = (const float*)d_in[11];
    const float* fa4_w  = (const float*)d_in[12];
    const float* fa4_b  = (const float*)d_in[13];
    const float* fa5_w  = (const float*)d_in[14];
    const float* fa5_b  = (const float*)d_in[15];
    const float* lp1_w  = (const float*)d_in[16];
    const float* lp1_b  = (const float*)d_in[17];
    const float* lp2_w  = (const float*)d_in[18];
    const float* lp2_b  = (const float*)d_in[19];
    const float* out_w  = (const float*)d_in[20];
    const float* out_b  = (const float*)d_in[21];

    float* s_out_f   = (float*)d_out;                       // [512,50,600]
    float* adj_out_f = (float*)d_out + (size_t)ROWS * 600;  // [512,50,50]

    // ---- workspace bump allocator (256B aligned) ----
    char* base = (char*)d_ws;
    size_t off = 0;
    auto alloc = [&](size_t bytes) -> void* {
        void* p = base + off;
        off = (off + bytes + 255) & ~(size_t)255;
        return p;
    };
    unsigned short* l_bf   = (unsigned short*)alloc((size_t)BATCH * LD * 2);
    unsigned short* w_fa1  = (unsigned short*)alloc((size_t)D1 * D1P * 2);
    unsigned short* w_fa2  = (unsigned short*)alloc((size_t)D1 * D1P * 2);
    unsigned short* w_fa3  = (unsigned short*)alloc((size_t)D2 * D2P * 2);
    unsigned short* w_fa4  = (unsigned short*)alloc((size_t)D2 * D2P * 2);
    unsigned short* w_fa5  = (unsigned short*)alloc((size_t)D2 * D2P * 2);
    unsigned short* w_lp1  = (unsigned short*)alloc((size_t)D2 * LD * 2);
    unsigned short* w_lp2  = (unsigned short*)alloc((size_t)D2 * LD * 2);
    unsigned short* w_out  = (unsigned short*)alloc((size_t)FEAT * D2P * 2);
    float*          lp1f   = (float*)alloc((size_t)BATCH * D2 * 4);
    float*          lp2f   = (float*)alloc((size_t)BATCH * D2 * 4);
    float*          adjraw = (float*)alloc((size_t)BATCH * NN * NN * 4);
    unsigned short* adjb   = (unsigned short*)alloc((size_t)BATCH * 4096 * 2);
    unsigned short* s_bb   = (unsigned short*)alloc((size_t)ROWS * D1P * 2);
    unsigned short* comb   = (unsigned short*)alloc((size_t)ROWS * D2P * 2);
    unsigned short* lms    = (unsigned short*)alloc((size_t)ROWS * D2P * 2);
    unsigned short* fa4r   = (unsigned short*)alloc((size_t)ROWS * D2P * 2);
    unsigned short* t1     = lms;    // fa1 scratch: lms written later
    unsigned short* t2     = fa4r;   // fa2 scratch: fa4r written later
    unsigned short* prep   = fa4r;   // fa4r dead after gemm_adj
    unsigned short* mp     = lms;    // lms  dead after gemm_adj

    auto cvt = [&](const float* src, unsigned short* dst, int rows, int K, int Kpad) {
        int total = rows * Kpad;
        cvt_kernel<<<(total + 255) / 256, 256, 0, stream>>>(src, dst, rows, K, Kpad);
    };
    cvt(l,     l_bf,  BATCH, LD, LD);
    cvt(fa1_w, w_fa1, D1,  D1, D1P);
    cvt(fa2_w, w_fa2, D1,  D1, D1P);
    cvt(fa3_w, w_fa3, D2,  D2, D2P);
    cvt(fa4_w, w_fa4, D2,  D2, D2P);
    cvt(fa5_w, w_fa5, D2,  D2, D2P);
    cvt(lp1_w, w_lp1, D2,  LD, LD);
    cvt(lp2_w, w_lp2, D2,  LD, LD);
    cvt(out_w, w_out, FEAT, D2, D2P);

    // bb features + concat(s, bb) -> s_bb (bf16), combined left half
    prep_kernel<<<ROWS, 256, 0, stream>>>(s, ps, bb_w, bb_b, s_bb, comb);

    auto blocks = [](int M, int N) { return (M >> 7) * ((N + 63) >> 6); };

    // fa1 / fa2 via LDS-staged GEMM, then elementwise product -> combined
    gemm_kernel<0><<<blocks(ROWS, D1), 256, 0, stream>>>(
        s_bb, D1P, w_fa1, D1P, fa1_b, ROWS, D1, D1P, nullptr, t1, D1P, 0, nullptr, nullptr);
    gemm_kernel<0><<<blocks(ROWS, D1), 256, 0, stream>>>(
        s_bb, D1P, w_fa2, D1P, fa2_b, ROWS, D1, D1P, nullptr, t2, D1P, 0, nullptr, nullptr);
    mul_kernel<<<(ROWS * D1 + 255) / 256, 256, 0, stream>>>(t1, t2, comb);

    // lp1 / lp2 : [512,2048] x [1200,2048]^T -> f32
    gemm_kernel<2><<<blocks(BATCH, D2), 256, 0, stream>>>(
        l_bf, LD, w_lp1, LD, lp1_b, BATCH, D2, LD, lp1f, nullptr, D2, 0, nullptr, nullptr);
    gemm_kernel<2><<<blocks(BATCH, D2), 256, 0, stream>>>(
        l_bf, LD, w_lp2, LD, lp2_b, BATCH, D2, LD, lp2f, nullptr, D2, 0, nullptr, nullptr);

    // lms = relu(combined@fa3^T + b) * lp1   (bf16, padded)
    gemm_kernel<1><<<blocks(ROWS, D2), 256, 0, stream>>>(
        comb, D2P, w_fa3, D2P, fa3_b, ROWS, D2, D2P, nullptr, lms, D2P, 0, lp1f, nullptr);

    // fa4r = relu(combined@fa4^T + b)        (bf16, padded)
    gemm_kernel<0><<<blocks(ROWS, D2), 256, 0, stream>>>(
        comb, D2P, w_fa4, D2P, fa4_b, ROWS, D2, D2P, nullptr, fa4r, D2P, 0, nullptr, nullptr);

    // adj_raw = fa4r . lms (per batch), softmax with mask
    gemm_adj_kernel<<<(BATCH * 4 + 7) / 8, 256, 0, stream>>>(fa4r, lms, adjraw);
    softmax_kernel<<<BATCH * 64, 64, 0, stream>>>(adjraw, mask_s, adj_out_f, adjb);

    // prepared = relu(combined@fa5^T + b) * lp2  (reuses fa4r buffer)
    gemm_kernel<1><<<blocks(ROWS, D2), 256, 0, stream>>>(
        comb, D2P, w_fa5, D2P, fa5_b, ROWS, D2, D2P, nullptr, prep, D2P, 0, lp2f, nullptr);

    // mp = adj @ prepared (per batch, reuses lms buffer)
    gemm_msg_kernel<<<(BATCH * 4 * 19 + 7) / 8, 256, 0, stream>>>(adjb, prep, mp);

    // messages = relu(mp@out_w^T + b) * mask -> d_out[:, :, 300:600]
    gemm_kernel<3><<<blocks(ROWS, FEAT), 256, 0, stream>>>(
        mp, D2P, w_out, D2P, out_b, ROWS, FEAT, D2P, s_out_f, nullptr, 2 * FEAT, FEAT,
        nullptr, mask_s);

    // s * mask -> d_out[:, :, 0:300]
    write_s_kernel<<<(ROWS * FEAT + 255) / 256, 256, 0, stream>>>(s, mask_s, s_out_f);
}
